// CustomWeightedGraphSAGE_72232759984605
// MI455X (gfx1250) — matrix operand order repaired and checked by reading, not compiled, not run on hardware
//
#include <hip/hip_runtime.h>
#include <hip/hip_bf16.h>

typedef _Float16 v16h __attribute__((ext_vector_type(16)));
typedef _Float16 v8h  __attribute__((ext_vector_type(8)));
typedef float    v8f  __attribute__((ext_vector_type(8)));

#define DIN   128          // feature dim of h, also D_OUT
#define DK    256          // GEMM K = 2*DIN
#define LDSK  (DK + 8)     // pad 8 halves (16B) -> row stride 132 dwords, 132%64=4 banks skew

// ---------------------------------------------------------------------------
// Kernel 1: zero the scatter accumulators and convert W (f32) -> Wh (f16).
// Must run every launch (workspace is not re-poisoned between graph replays).
// ---------------------------------------------------------------------------
__global__ void sage_init(float* __restrict__ h_sum, float* __restrict__ deg,
                          _Float16* __restrict__ Wh, const float* __restrict__ W,
                          int n_hsum, int n_deg, int n_w) {
    const int total = n_hsum + n_deg + n_w;
    for (int idx = blockIdx.x * blockDim.x + threadIdx.x; idx < total;
         idx += gridDim.x * blockDim.x) {
        if (idx < n_hsum) {
            h_sum[idx] = 0.0f;
        } else if (idx < n_hsum + n_deg) {
            deg[idx - n_hsum] = 0.0f;
        } else {
            const int i = idx - n_hsum - n_deg;
            Wh[i] = (_Float16)W[i];
        }
    }
}

// ---------------------------------------------------------------------------
// Kernel 2: edge scatter. One thread per (edge, feature). h_sum fits in the
// 192MB L2, so the f32 atomics stay on-chip. Consecutive threads hit
// consecutive features -> coalesced gather of h[src] and clustered atomics.
// ---------------------------------------------------------------------------
__global__ void sage_scatter(const float* __restrict__ h, const float* __restrict__ w,
                             const int* __restrict__ src, const int* __restrict__ dst,
                             float* __restrict__ h_sum, float* __restrict__ deg,
                             long long total) {
    long long idx = (long long)blockIdx.x * blockDim.x + threadIdx.x;
    if (idx >= total) return;
    const int e = (int)(idx >> 7);        // DIN == 128
    const int f = (int)(idx & (DIN - 1));
    const int s = src[e];
    const int d = dst[e];
    const float val = h[(size_t)s * DIN + f] * w[e];
    atomicAdd(&h_sum[(size_t)d * DIN + f], val);
    if (f == 0) atomicAdd(&deg[d], 1.0f);
}

// ---------------------------------------------------------------------------
// Kernel 3: fused concat + mean-normalize + GEMM(+bias) using WMMA f16->f32.
//   out[m][n] = sum_k htot[m][k] * W[n][k] + b[n]
// Block = 256 threads = 8 wave32 waves; block owns a 16-row M tile, wave w
// owns N columns [16w, 16w+16). K loop: 8 steps of 32 (v_wmma_f32_16x16x32_f16).
// ---------------------------------------------------------------------------
__global__ __launch_bounds__(256) void sage_gemm(
    const float* __restrict__ h, const float* __restrict__ h_sum,
    const float* __restrict__ deg, const _Float16* __restrict__ Wh,
    const float* __restrict__ bias, float* __restrict__ out) {

    __shared__ _Float16 Atile[16][LDSK];

    const int t  = threadIdx.x;
    const int m0 = blockIdx.x * 16;

    // Stage A tile (16 x 256 f16): cols 0..127 = h, cols 128..255 = h_sum/max(deg,1).
    // 4096 elems / 256 threads = 16 each; consecutive t -> consecutive cols (coalesced).
    for (int idx = t; idx < 16 * DK; idx += 256) {
        const int row  = idx >> 8;      // DK == 256
        const int col  = idx & (DK - 1);
        const int grow = m0 + row;
        float v;
        if (col < DIN) {
            v = h[(size_t)grow * DIN + col];
        } else {
            float d = deg[grow];
            d = (d > 1.0f) ? d : 1.0f;
            v = h_sum[(size_t)grow * DIN + (col - DIN)] / d;
        }
        Atile[row][col] = (_Float16)v;
    }
    __syncthreads();

    const int wave = t >> 5;
    const int lane = t & 31;
    const int n    = lane & 15;   // also the A-row index m for this lane
    const int hi   = lane >> 4;
    const int n0   = wave * 16;

    const _Float16* wrow = Wh + (size_t)(n0 + n) * DK;  // B lane reads one W row: unit stride

    v8f acc = {};
#pragma unroll
    for (int kb = 0; kb < DK; kb += 32) {
        // A fragment (16x32, MxK): lane m holds K = {hi*8..hi*8+7} U {16+hi*8..16+hi*8+7}
        v8h a0 = *(const v8h*)&Atile[n][kb + hi * 8];
        v8h a1 = *(const v8h*)&Atile[n][kb + 16 + hi * 8];
        // B fragment (32x16, KxN): lane n holds K = {hi*8..hi*8+7} U {16+hi*8..16+hi*8+7}, same as A
        v8h b0 = *(const v8h*)(wrow + kb + hi * 8);
        v8h b1 = *(const v8h*)(wrow + kb + 16 + hi * 8);
        v16h afrag, bfrag;
#pragma unroll
        for (int i = 0; i < 8; ++i) {
            afrag[i] = a0[i]; afrag[i + 8] = a1[i];
            bfrag[i] = b0[i]; bfrag[i + 8] = b1[i];
        }
        // (neg_a, A, neg_b, B, c_mod, C, reuse_a, reuse_b)
        acc = __builtin_amdgcn_wmma_f32_16x16x32_f16(
            false, afrag, false, bfrag, (short)0, acc, false, false);
    }

    // C/D layout: VGPR r -> M = r + hi*8, N = lane&15 (within tile)
    const float bv    = bias[n0 + n];
    const int   rbase = m0 + hi * 8;
#pragma unroll
    for (int r = 0; r < 8; ++r) {
        out[(size_t)(rbase + r) * DIN + n0 + n] = acc[r] + bv;
    }
}

// ---------------------------------------------------------------------------
// Launch. Input order: h, w, src, dst, W, b. Indices arrive as int32
// (JAX without x64 demotes the int64 randint; harness doc: integer -> int*).
// Workspace layout: h_sum [N*128 f32] | deg [N f32] | Wh [128*256 f16]
// (offset of Wh = N*129*4 = 51,600,000 bytes, 16B-aligned for b128 loads).
// ---------------------------------------------------------------------------
extern "C" void kernel_launch(void* const* d_in, const int* in_sizes, int n_in,
                              void* d_out, int out_size, void* d_ws, size_t ws_size,
                              hipStream_t stream) {
    const float* h   = (const float*)d_in[0];
    const float* w   = (const float*)d_in[1];
    const int*   src = (const int*)d_in[2];
    const int*   dst = (const int*)d_in[3];
    const float* W   = (const float*)d_in[4];
    const float* b   = (const float*)d_in[5];
    float*       out = (float*)d_out;

    const int N = in_sizes[0] / DIN;   // 100000
    const int E = in_sizes[2];         // 640000

    float*    h_sum = (float*)d_ws;
    float*    deg   = h_sum + (size_t)N * DIN;
    _Float16* Wh    = (_Float16*)(deg + N);

    const int n_hsum = N * DIN;
    const int n_w    = DIN * DK;

    sage_init<<<2048, 256, 0, stream>>>(h_sum, deg, Wh, W, n_hsum, N, n_w);

    const long long total = (long long)E * DIN;
    const unsigned  sblocks = (unsigned)((total + 255) / 256);
    sage_scatter<<<sblocks, 256, 0, stream>>>(h, w, src, dst, h_sum, deg, total);

    sage_gemm<<<N / 16, 256, 0, stream>>>(h, h_sum, deg, Wh, b, out);
}
